// MuseformerDecoderLayer_34050500723067
// MI455X (gfx1250) — matrix operand order repaired and checked by reading, not compiled, not run on hardware
//
#include <hip/hip_runtime.h>
#include <hip/hip_bf16.h>

// ---------------------------------------------------------------------------
// Museformer decoder layer for MI455X (gfx1250, wave32, WMMA bf16 path).
// B=1, L=4096, DM=1024, H=16, DH=64, BS=64, W=16, MAXD=256, DFF=4096, NB=64
// - all matmuls on v_wmma_f32_16x16x32_bf16 (f32 accum)
// - B operands pre-transposed ([N][K]) so fragment loads are 2x ds_load_b128
// - global->LDS via async-DMA (GLOBAL_LOAD_ASYNC_TO_LDS_B128 / ASYNCcnt)
//   with double-buffered software pipeline in the GEMM
// ---------------------------------------------------------------------------

#define L_    4096
#define DM_   1024
#define H_    16
#define DH_   64
#define BS_   64
#define W_    16
#define MAXD_ 256
#define DFF_  4096
#define NB_   64

typedef __attribute__((ext_vector_type(16))) __bf16 v16bf;
typedef __attribute__((ext_vector_type(8)))  float  v8f;
typedef int v4i __attribute__((vector_size(16)));   // matches async-LDS builtin param

union BF16x16 { v16bf v; uint4 q[2]; };

#if defined(__has_builtin)
#if __has_builtin(__builtin_amdgcn_global_load_async_to_lds_b128) && \
    __has_builtin(__builtin_amdgcn_s_wait_asynccnt)
#define USE_ASYNC_LDS 1
#endif
#endif

// ----------------------------- helpers -------------------------------------

static __device__ __forceinline__ __bf16 f2bf(float f) {
    union { float f; unsigned u; } a; a.f = f;
    unsigned u = a.u;
    unsigned r = (u + 0x7FFFu + ((u >> 16) & 1u)) >> 16;
    union { unsigned short s; __bf16 b; } o; o.s = (unsigned short)r;
    return o.b;
}

#ifdef USE_ASYNC_LDS
// async-copy 16 bf16 (32B) from global to LDS: two b128 DMA ops
static __device__ __forceinline__ void async_cp32B(__bf16* ldsDst, const __bf16* gSrc) {
    __builtin_amdgcn_global_load_async_to_lds_b128((v4i*)gSrc,       (v4i*)ldsDst,       0, 0);
    __builtin_amdgcn_global_load_async_to_lds_b128((v4i*)(gSrc + 8), (v4i*)(ldsDst + 8), 0, 0);
}
// async-copy 32 bf16 (64B)
static __device__ __forceinline__ void async_cp64B(__bf16* ldsDst, const __bf16* gSrc) {
    async_cp32B(ldsDst,      gSrc);
    async_cp32B(ldsDst + 16, gSrc + 16);
}
#endif

// A-fragment (16x32 bf16, M x K), contiguous K layout (matches load_bt_frag):
// lanes 0-15 : M=lane,    K = 0..15
// lanes 16-31: M=lane-16, K = 16..31
static __device__ __forceinline__ v16bf load_a_frag(const __bf16* lds, int rowBase,
                                                    int ld, int kBase) {
    int lane = threadIdx.x & 31;
    int m  = lane & 15;
    int kb = (lane >> 4) ? 16 : 0;
    const uint4* p = (const uint4*)(lds + (size_t)(rowBase + m) * ld + kBase + kb);
    BF16x16 u;
    u.q[0] = p[0];   // K = kb+0..7
    u.q[1] = p[1];   // K = kb+8..15
    return u.v;
}

// B-fragment (32x16 bf16, K x N) from an [N][K] (transposed) LDS tile:
// lanes 0-15 : N=lane,    K = 0..15 ; lanes 16-31: N=lane-16, K = 16..31
static __device__ __forceinline__ v16bf load_bt_frag(const __bf16* ldsT, int colBase,
                                                     int ld, int kBase) {
    int lane = threadIdx.x & 31;
    int n  = lane & 15;
    int kb = (lane >> 4) ? 16 : 0;
    const uint4* p = (const uint4*)(ldsT + (size_t)(colBase + n) * ld + kBase + kb);
    BF16x16 u;
    u.q[0] = p[0];
    u.q[1] = p[1];
    return u.v;
}

static __device__ __forceinline__ float gelu_tanh(float v) {
    float t = 0.7978845608028654f * (v + 0.044715f * v * v * v);
    return 0.5f * v * (1.0f + tanhf(t));
}

// --------------- f32 -> bf16 convert WITH transpose (weights) ---------------
// in: [K][N] f32 row-major   out: [N][K] bf16 row-major

__global__ __launch_bounds__(256)
void cvt_t_bf16_kernel(const float* __restrict__ in, __bf16* __restrict__ out,
                       int K, int N) {
    int i = blockIdx.x * 256 + threadIdx.x;
    if (i < K * N) {
        int n = i / K, k = i - n * K;
        out[i] = f2bf(in[(size_t)k * N + n]);
    }
}

// ------------------------------ LayerNorm -----------------------------------

__global__ __launch_bounds__(256)
void ln_bf16_kernel(const float* __restrict__ x, const float* __restrict__ g,
                    const float* __restrict__ b, __bf16* __restrict__ out) {
    __shared__ float red[256];
    const int row = blockIdx.x;
    const int tid = threadIdx.x;
    const float* xr = x + (size_t)row * DM_;

    float s = 0.f;
    for (int i = tid; i < DM_; i += 256) s += xr[i];
    red[tid] = s; __syncthreads();
    for (int o = 128; o > 0; o >>= 1) { if (tid < o) red[tid] += red[tid + o]; __syncthreads(); }
    const float mean = red[0] * (1.0f / DM_);
    __syncthreads();

    float v = 0.f;
    for (int i = tid; i < DM_; i += 256) { float d = xr[i] - mean; v += d * d; }
    red[tid] = v; __syncthreads();
    for (int o = 128; o > 0; o >>= 1) { if (tid < o) red[tid] += red[tid + o]; __syncthreads(); }
    const float inv = rsqrtf(red[0] * (1.0f / DM_) + 1e-5f);

    __bf16* orow = out + (size_t)row * DM_;
    for (int i = tid; i < DM_; i += 256)
        orow[i] = f2bf((xr[i] - mean) * inv * g[i] + b[i]);
}

// ------------------------- generic bf16 WMMA GEMM ---------------------------
// C[M x N] = A[M x K] * B[K x N], with B supplied PRE-TRANSPOSED: Bt[N x K].
// Tile 64x64, BK=32, 128 threads (4 waves). Double-buffered LDS pipeline:
// issue next tile's async DMA, compute WMMAs on current, wait+barrier, swap.

__global__ __launch_bounds__(128)
void gemm_bf16_kernel(const __bf16* __restrict__ A, const __bf16* __restrict__ Bt,
                      int M, int N, int K,
                      const float* __restrict__ bias,       // length N, nullable
                      const float* __restrict__ residual,   // M x N f32, nullable
                      int apply_gelu,
                      float* __restrict__ outF,             // nullable, (M,N)
                      __bf16* __restrict__ outB,            // nullable
                      int outB_trans) {                     // 0: (M,N), 1: (N,M)
    __shared__ __align__(16) __bf16 As[2][64 * 32];   // [M][K]
    __shared__ __align__(16) __bf16 BsT[2][64 * 32];  // [N][K]

    const int t    = threadIdx.x;
    const int lane = t & 31;
    const int wv   = t >> 5;
    const int rowB = wv * 16;
    const int bm   = blockIdx.y * 64;
    const int bn   = blockIdx.x * 64;

    const int r2  = t >> 1;          // 0..63
    const int sg2 = (t & 1) * 16;    // 0 or 16

    const __bf16* gA = A  + (size_t)(bm + r2) * K + sg2;  // advance by kt
    const __bf16* gB = Bt + (size_t)(bn + r2) * K + sg2;
    __bf16* const dA0 = &As[0][r2 * 32 + sg2];
    __bf16* const dA1 = &As[1][r2 * 32 + sg2];
    __bf16* const dB0 = &BsT[0][r2 * 32 + sg2];
    __bf16* const dB1 = &BsT[1][r2 * 32 + sg2];

    v8f acc0 = {}, acc1 = {}, acc2 = {}, acc3 = {};

#ifdef USE_ASYNC_LDS
    // prologue: DMA tile 0 into buffer 0
    async_cp32B(dA0, gA);
    async_cp32B(dB0, gB);
    __builtin_amdgcn_s_wait_asynccnt(0);
    __syncthreads();

    int cur = 0;
    for (int kt = 0; kt < K; kt += 32) {
        const int nxt = cur ^ 1;
        const bool hasNext = (kt + 32 < K);
        if (hasNext) {  // fire-and-forget DMA of next tile, overlaps the WMMAs
            async_cp32B(nxt ? dA1 : dA0, gA + kt + 32);
            async_cp32B(nxt ? dB1 : dB0, gB + kt + 32);
        }
        const __bf16* as = As[cur];
        const __bf16* bs = BsT[cur];
        v16bf a  = load_a_frag(as, rowB, 32, 0);
        v16bf b0 = load_bt_frag(bs,  0, 32, 0);
        v16bf b1 = load_bt_frag(bs, 16, 32, 0);
        v16bf b2 = load_bt_frag(bs, 32, 32, 0);
        v16bf b3 = load_bt_frag(bs, 48, 32, 0);
        acc0 = __builtin_amdgcn_wmma_f32_16x16x32_bf16(false, a, false, b0, (short)0, acc0, false, false);
        acc1 = __builtin_amdgcn_wmma_f32_16x16x32_bf16(false, a, false, b1, (short)0, acc1, false, false);
        acc2 = __builtin_amdgcn_wmma_f32_16x16x32_bf16(false, a, false, b2, (short)0, acc2, false, false);
        acc3 = __builtin_amdgcn_wmma_f32_16x16x32_bf16(false, a, false, b3, (short)0, acc3, false, false);
        __builtin_amdgcn_s_wait_asynccnt(0);
        __syncthreads();
        cur = nxt;
    }
#else
    // staged-register pipeline fallback: issue loads, compute, commit stores
    uint4 ra0, ra1, rb0, rb1;
    {
        const uint4* sa = (const uint4*)(gA);
        const uint4* sb = (const uint4*)(gB);
        ra0 = sa[0]; ra1 = sa[1]; rb0 = sb[0]; rb1 = sb[1];
        uint4* da = (uint4*)dA0; uint4* db = (uint4*)dB0;
        da[0] = ra0; da[1] = ra1; db[0] = rb0; db[1] = rb1;
    }
    __syncthreads();

    int cur = 0;
    for (int kt = 0; kt < K; kt += 32) {
        const int nxt = cur ^ 1;
        const bool hasNext = (kt + 32 < K);
        if (hasNext) {   // issue next tile's global loads (latency hidden by WMMAs)
            const uint4* sa = (const uint4*)(gA + kt + 32);
            const uint4* sb = (const uint4*)(gB + kt + 32);
            ra0 = sa[0]; ra1 = sa[1]; rb0 = sb[0]; rb1 = sb[1];
            __builtin_prefetch(gA + kt + 64, 0, 3);
            __builtin_prefetch(gB + kt + 64, 0, 3);
        }
        const __bf16* as = As[cur];
        const __bf16* bs = BsT[cur];
        v16bf a  = load_a_frag(as, rowB, 32, 0);
        v16bf b0 = load_bt_frag(bs,  0, 32, 0);
        v16bf b1 = load_bt_frag(bs, 16, 32, 0);
        v16bf b2 = load_bt_frag(bs, 32, 32, 0);
        v16bf b3 = load_bt_frag(bs, 48, 32, 0);
        acc0 = __builtin_amdgcn_wmma_f32_16x16x32_bf16(false, a, false, b0, (short)0, acc0, false, false);
        acc1 = __builtin_amdgcn_wmma_f32_16x16x32_bf16(false, a, false, b1, (short)0, acc1, false, false);
        acc2 = __builtin_amdgcn_wmma_f32_16x16x32_bf16(false, a, false, b2, (short)0, acc2, false, false);
        acc3 = __builtin_amdgcn_wmma_f32_16x16x32_bf16(false, a, false, b3, (short)0, acc3, false, false);
        if (hasNext) {
            uint4* da = (uint4*)(nxt ? dA1 : dA0);
            uint4* db = (uint4*)(nxt ? dB1 : dB0);
            da[0] = ra0; da[1] = ra1; db[0] = rb0; db[1] = rb1;
        }
        __syncthreads();
        cur = nxt;
    }
#endif

    // C element (vgpr r, lane): row = rowB + r + 8*(lane>=16), col = f*16 + (lane&15)
    const int rofs = (lane >> 4) * 8;
    const int cl   = lane & 15;
    v8f accs[4] = {acc0, acc1, acc2, acc3};
#pragma unroll
    for (int f = 0; f < 4; ++f) {
#pragma unroll
        for (int r = 0; r < 8; ++r) {
            int gr = bm + rowB + r + rofs;
            int gc = bn + f * 16 + cl;
            float v = accs[f][r];
            if (bias)       v += bias[gc];
            if (apply_gelu) v  = gelu_tanh(v);
            if (residual)   v += residual[(size_t)gr * N + gc];
            if (outF) outF[(size_t)gr * N + gc] = v;
            if (outB) {
                if (outB_trans) outB[(size_t)gc * M + gr] = f2bf(v);
                else            outB[(size_t)gr * N + gc] = f2bf(v);
            }
        }
    }
}

// ---------------------- banded attention (flash-style) ----------------------
// One workgroup per (query block n, head h). 128 threads = 4 waves.
// q,k: (L, DM) bf16;  vT: (DM, L) bf16 (pre-transposed by the V projection).

__global__ __launch_bounds__(128)
void attn_kernel(const __bf16* __restrict__ q,
                 const __bf16* __restrict__ k,
                 const __bf16* __restrict__ vT,
                 const float*  __restrict__ rel_emb,   // (MAXD+2, H) f32
                 __bf16* __restrict__ o) {             // (L, DM) bf16
    __shared__ __align__(16) __bf16 Qs[BS_ * DH_];    // [i][d]   (A for QK^T)
    __shared__ __align__(16) __bf16 Ks[BS_ * DH_];    // [j][d]   ([N][K] for QK^T)
    __shared__ __align__(16) __bf16 VsT[DH_ * BS_];   // [d][j]   ([N][K] for PV)
    __shared__ __align__(16) float  Sf[BS_ * BS_];    // scores
    __shared__ __align__(16) __bf16 Pb[BS_ * BS_];    // probabilities (A for PV)
    __shared__ __align__(16) float  Of[BS_ * DH_];    // output accumulator
    __shared__ float row_m[BS_], row_s[BS_], row_a[BS_];

    const int n = blockIdx.x;
    const int h = blockIdx.y;
    const int t = threadIdx.x;
    const int lane = t & 31;
    const int wv   = t >> 5;
    const int rowB = wv * 16;
    const int rofs = (lane >> 4) * 8;
    const int cl   = lane & 15;

    const int r2  = t >> 1;          // 0..63
    const int sg2 = (t & 1) * 32;    // 0 or 32

    // Q tile (row-major, contiguous)
    {
        const uint4* s = (const uint4*)(q + (size_t)(n * BS_ + r2) * DM_ + h * DH_ + sg2);
        uint4* d = (uint4*)(&Qs[r2 * DH_ + sg2]);
        d[0] = s[0]; d[1] = s[1]; d[2] = s[2]; d[3] = s[3];
    }
    for (int idx = t; idx < BS_ * DH_; idx += 128) Of[idx] = 0.f;
    if (t < BS_) { row_m[t] = -INFINITY; row_s[t] = 0.f; }
    __syncthreads();

    for (int w = 0; w < W_; ++w) {
        const int src_blk = n - (W_ - 1) + w;
        const int src_idx = src_blk < 0 ? 0 : src_blk;

        const __bf16* gK = k  + (size_t)(src_idx * BS_ + r2) * DM_ + h * DH_ + sg2;
        const __bf16* gV = vT + (size_t)(h * DH_ + r2) * L_ + src_idx * BS_ + sg2;
#ifdef USE_ASYNC_LDS
        async_cp64B(&Ks[r2 * DH_ + sg2],  gK);
        async_cp64B(&VsT[r2 * BS_ + sg2], gV);
        __builtin_amdgcn_s_wait_asynccnt(0);
        __syncthreads();
#else
        {
            const uint4* s = (const uint4*)gK;
            uint4* d = (uint4*)(&Ks[r2 * DH_ + sg2]);
            uint4 t0 = s[0], t1 = s[1], t2 = s[2], t3 = s[3];
            d[0] = t0; d[1] = t1; d[2] = t2; d[3] = t3;
        }
        {
            const uint4* s = (const uint4*)gV;
            uint4* d = (uint4*)(&VsT[r2 * BS_ + sg2]);
            uint4 t0 = s[0], t1 = s[1], t2 = s[2], t3 = s[3];
            d[0] = t0; d[1] = t1; d[2] = t2; d[3] = t3;
        }
        __syncthreads();
#endif

        // S = Q * K^T  (two K-steps of 32 over DH=64)
        v8f s0 = {}, s1 = {}, s2 = {}, s3 = {};
#pragma unroll
        for (int kk = 0; kk < DH_; kk += 32) {
            v16bf a  = load_a_frag(Qs, rowB, DH_, kk);
            v16bf b0 = load_bt_frag(Ks,  0, DH_, kk);
            v16bf b1 = load_bt_frag(Ks, 16, DH_, kk);
            v16bf b2 = load_bt_frag(Ks, 32, DH_, kk);
            v16bf b3 = load_bt_frag(Ks, 48, DH_, kk);
            s0 = __builtin_amdgcn_wmma_f32_16x16x32_bf16(false, a, false, b0, (short)0, s0, false, false);
            s1 = __builtin_amdgcn_wmma_f32_16x16x32_bf16(false, a, false, b1, (short)0, s1, false, false);
            s2 = __builtin_amdgcn_wmma_f32_16x16x32_bf16(false, a, false, b2, (short)0, s2, false, false);
            s3 = __builtin_amdgcn_wmma_f32_16x16x32_bf16(false, a, false, b3, (short)0, s3, false, false);
        }
        // scale + relative bias + mask -> Sf
        {
            v8f ss[4] = {s0, s1, s2, s3};
#pragma unroll
            for (int f = 0; f < 4; ++f) {
#pragma unroll
                for (int r = 0; r < 8; ++r) {
                    int i = rowB + r + rofs;
                    int j = f * 16 + cl;
                    bool valid = (src_blk >= 0) && ((w < W_ - 1) || (j <= i));
                    float sc = ss[f][r] * 0.125f;   // 1/sqrt(DH)
                    int rel = (n * BS_ + i) - (src_idx * BS_ + j);
                    rel = rel < -1 ? -1 : (rel > MAXD_ ? MAXD_ : rel);
                    sc += rel_emb[(rel + 1) * H_ + h];
                    Sf[i * BS_ + j] = valid ? sc : -1e30f;
                }
            }
        }
        __syncthreads();

        // online softmax update (one thread per row)
        if (t < BS_) {
            const int i = t;
            float mloc = -1e30f;
            for (int j = 0; j < BS_; ++j) mloc = fmaxf(mloc, Sf[i * BS_ + j]);
            float mold = row_m[i];
            float mnew = fmaxf(mold, mloc);
            float alpha = __expf(mold - mnew);
            float ssum = 0.f;
            for (int j = 0; j < BS_; ++j) {
                float p = __expf(Sf[i * BS_ + j] - mnew);
                Pb[i * BS_ + j] = f2bf(p);
                ssum += p;
            }
            row_a[i] = alpha;
            row_m[i] = mnew;
            row_s[i] = row_s[i] * alpha + ssum;
        }
        __syncthreads();

        // rescale O accumulator
        for (int idx = t; idx < BS_ * DH_; idx += 128) Of[idx] *= row_a[idx >> 6];
        __syncthreads();

        // O += P * V   (A = Pb [i][j], B = VsT [d][j] == [N][K])
        v8f p0 = {}, p1 = {}, p2 = {}, p3 = {};
#pragma unroll
        for (int kk = 0; kk < BS_; kk += 32) {
            v16bf a  = load_a_frag(Pb, rowB, BS_, kk);
            v16bf b0 = load_bt_frag(VsT,  0, BS_, kk);
            v16bf b1 = load_bt_frag(VsT, 16, BS_, kk);
            v16bf b2 = load_bt_frag(VsT, 32, BS_, kk);
            v16bf b3 = load_bt_frag(VsT, 48, BS_, kk);
            p0 = __builtin_amdgcn_wmma_f32_16x16x32_bf16(false, a, false, b0, (short)0, p0, false, false);
            p1 = __builtin_amdgcn_wmma_f32_16x16x32_bf16(false, a, false, b1, (short)0, p1, false, false);
            p2 = __builtin_amdgcn_wmma_f32_16x16x32_bf16(false, a, false, b2, (short)0, p2, false, false);
            p3 = __builtin_amdgcn_wmma_f32_16x16x32_bf16(false, a, false, b3, (short)0, p3, false, false);
        }
        {
            v8f pp[4] = {p0, p1, p2, p3};
#pragma unroll
            for (int f = 0; f < 4; ++f)
#pragma unroll
                for (int r = 0; r < 8; ++r) {
                    int i = rowB + r + rofs;
                    int d = f * 16 + cl;
                    Of[i * DH_ + d] += pp[f][r];
                }
        }
        __syncthreads();
    }

    // normalize and write o in (L, DM) layout
    if (t < BS_) {
        const int i = t;
        float inv = 1.0f / row_s[i];
        __bf16* orow = o + (size_t)(n * BS_ + i) * DM_ + h * DH_;
        for (int d = 0; d < DH_; ++d) orow[d] = f2bf(Of[i * DH_ + d] * inv);
    }
}

// ------------------------------- launcher -----------------------------------

extern "C" void kernel_launch(void* const* d_in, const int* in_sizes, int n_in,
                              void* d_out, int out_size, void* d_ws, size_t ws_size,
                              hipStream_t stream) {
    const float* x      = (const float*)d_in[0];
    const float* Wq     = (const float*)d_in[1];
    const float* Wk     = (const float*)d_in[2];
    const float* Wv     = (const float*)d_in[3];
    const float* Wo     = (const float*)d_in[4];
    const float* rel    = (const float*)d_in[5];
    const float* ln1_g  = (const float*)d_in[6];
    const float* ln1_b  = (const float*)d_in[7];
    const float* ln2_g  = (const float*)d_in[8];
    const float* ln2_b  = (const float*)d_in[9];
    const float* W1     = (const float*)d_in[10];
    const float* b1     = (const float*)d_in[11];
    const float* W2     = (const float*)d_in[12];
    const float* b2     = (const float*)d_in[13];
    float* out = (float*)d_out;

    char* ws = (char*)d_ws;
    size_t ofs = 0;
    auto carve = [&](size_t bytes) { char* p = ws + ofs; ofs += (bytes + 255) & ~(size_t)255; return p; };

    // all weight copies are TRANSPOSED: [N][K] bf16
    __bf16* WqT  = (__bf16*)carve((size_t)DM_ * DM_  * 2);
    __bf16* WkT  = (__bf16*)carve((size_t)DM_ * DM_  * 2);
    __bf16* WvT  = (__bf16*)carve((size_t)DM_ * DM_  * 2);
    __bf16* WoT  = (__bf16*)carve((size_t)DM_ * DM_  * 2);
    __bf16* W1T  = (__bf16*)carve((size_t)DM_ * DFF_ * 2);
    __bf16* W2T  = (__bf16*)carve((size_t)DFF_ * DM_ * 2);
    __bf16* h_bf   = (__bf16*)carve((size_t)L_ * DM_  * 2);
    __bf16* q_bf   = (__bf16*)carve((size_t)L_ * DM_  * 2);
    __bf16* k_bf   = (__bf16*)carve((size_t)L_ * DM_  * 2);
    __bf16* vT_bf  = (__bf16*)carve((size_t)L_ * DM_  * 2);  // (DM, L)
    __bf16* o_bf   = (__bf16*)carve((size_t)L_ * DM_  * 2);
    __bf16* h2_bf  = (__bf16*)carve((size_t)L_ * DM_  * 2);
    __bf16* act_bf = (__bf16*)carve((size_t)L_ * DFF_ * 2);
    float*  x1     = (float*) carve((size_t)L_ * DM_  * 4);

    // 1) convert + transpose weights to bf16
    auto cvt_t = [&](const float* src, __bf16* dst, int K, int N) {
        int n = K * N;
        cvt_t_bf16_kernel<<<(n + 255) / 256, 256, 0, stream>>>(src, dst, K, N);
    };
    cvt_t(Wq, WqT, DM_, DM_);
    cvt_t(Wk, WkT, DM_, DM_);
    cvt_t(Wv, WvT, DM_, DM_);
    cvt_t(Wo, WoT, DM_, DM_);
    cvt_t(W1, W1T, DM_, DFF_);
    cvt_t(W2, W2T, DFF_, DM_);

    // 2) LN1
    ln_bf16_kernel<<<L_, 256, 0, stream>>>(x, ln1_g, ln1_b, h_bf);

    // 3) Q/K/V projections (V written transposed for the attention PV matmul)
    dim3 gProj(DM_ / 64, L_ / 64);
    gemm_bf16_kernel<<<gProj, 128, 0, stream>>>(h_bf, WqT, L_, DM_, DM_,
                                                nullptr, nullptr, 0, nullptr, q_bf, 0);
    gemm_bf16_kernel<<<gProj, 128, 0, stream>>>(h_bf, WkT, L_, DM_, DM_,
                                                nullptr, nullptr, 0, nullptr, k_bf, 0);
    gemm_bf16_kernel<<<gProj, 128, 0, stream>>>(h_bf, WvT, L_, DM_, DM_,
                                                nullptr, nullptr, 0, nullptr, vT_bf, 1);

    // 4) banded attention
    attn_kernel<<<dim3(NB_, H_), 128, 0, stream>>>(q_bf, k_bf, vT_bf, rel, o_bf);

    // 5) output projection + residual -> x1 (f32)
    gemm_bf16_kernel<<<gProj, 128, 0, stream>>>(o_bf, WoT, L_, DM_, DM_,
                                                nullptr, x, 0, x1, nullptr, 0);

    // 6) LN2
    ln_bf16_kernel<<<L_, 256, 0, stream>>>(x1, ln2_g, ln2_b, h2_bf);

    // 7) FFN up + GELU
    dim3 gUp(DFF_ / 64, L_ / 64);
    gemm_bf16_kernel<<<gUp, 128, 0, stream>>>(h2_bf, W1T, L_, DFF_, DM_,
                                              b1, nullptr, 1, nullptr, act_bf, 0);

    // 8) FFN down + bias + residual -> out (f32)
    gemm_bf16_kernel<<<gProj, 128, 0, stream>>>(act_bf, W2T, L_, DM_, DFF_,
                                                b2, x1, 0, out, nullptr, 0);
}
